// MultiHeadAttention_5360119185455
// MI455X (gfx1250) — matrix operand order repaired and checked by reading, not compiled, not run on hardware
//
#include <hip/hip_runtime.h>

// ---------------------------------------------------------------------------
// MI455X (gfx1250) fused MHA: bf16 WMMA GEMMs + flash-style attention.
// B=4, S=2048, D=1024, H=16, HD=64.  Wave32, V_WMMA_F32_16X16X32_BF16,
// double-buffered GLOBAL_LOAD_ASYNC_TO_LDS_B128 pipelines (ASYNCcnt
// threshold waits, peeled epilogue), packed bf16 conversions, prefetch.
// ---------------------------------------------------------------------------

#define B_  4
#define S_  2048
#define D_  1024
#define H_  16
#define HD_ 64
#define M_  (B_ * S_)          // 8192 rows

typedef __attribute__((ext_vector_type(16))) __bf16 bf16x16;
typedef __attribute__((ext_vector_type(8)))  float  f32x8;
typedef __attribute__((ext_vector_type(4)))  float  f32x4v;
typedef __attribute__((ext_vector_type(2)))  float  f32x2v;
typedef __attribute__((ext_vector_type(4)))  __bf16 bf16x4;
typedef __attribute__((ext_vector_type(2)))  __bf16 bf16x2;
typedef __attribute__((ext_vector_type(4)))  int    i32x4;

#define AS1 __attribute__((address_space(1)))
#define AS3 __attribute__((address_space(3)))

union BFrag { unsigned int u[8]; bf16x16 v; };

// Packed f32 -> bf16 conversions (backend selects v_cvt_pk_bf16_f32).
__device__ __forceinline__ unsigned int pack2_bf16(float a, float b) {
    f32x2v f = {a, b};
    bf16x2 h = __builtin_convertvector(f, bf16x2);
    union { bf16x2 v; unsigned int u; } cv; cv.v = h;
    return cv.u;
}
__device__ __forceinline__ unsigned long long pack4_bf16(float4 f) {
    f32x4v v = {f.x, f.y, f.z, f.w};
    bf16x4 h = __builtin_convertvector(v, bf16x4);
    union { bf16x4 v; unsigned long long u; } cv; cv.v = h;
    return cv.u;
}
__device__ __forceinline__ unsigned short cvt1_bf16(float a) {
    union { bf16x2 v; unsigned short u[2]; } cv;
    cv.v = __builtin_convertvector((f32x2v){a, a}, bf16x2);
    return cv.u[0];
}

// A-fragment K-pair base for 16-bit 16x32 A matrix (ISA 7.12.2):
// half 0 lanes hold K {0..7,16..23}, half 1 lanes hold K {8..15,24..31}.
__device__ __forceinline__ int kpair(int j, int half) {
    return ((j & 4) << 2) + half * 8 + ((j & 3) << 1);
}

__device__ __forceinline__ f32x8 wmma_bf16(const BFrag& a, const BFrag& b, f32x8 c) {
    return __builtin_amdgcn_wmma_f32_16x16x32_bf16(
        false, a.v, false, b.v, (short)0, c, false, false);
}

// ---------------------------------------------------------------------------
// 16-byte global->LDS copy: async DMA path (ASYNCcnt), signature
// (AS1 int4*, AS3 int4*, imm offset, cpol) — confirmed by round-2 probe.
// ---------------------------------------------------------------------------
#if defined(__AMDGCN__) && __has_builtin(__builtin_amdgcn_global_load_async_to_lds_b128)
#define HAVE_ASYNC_LDS 1
#else
#define HAVE_ASYNC_LDS 0
#endif

__device__ __forceinline__ void cp16_g2s(unsigned short* l, const unsigned short* g) {
#if HAVE_ASYNC_LDS
    __builtin_amdgcn_global_load_async_to_lds_b128(
        (AS1 i32x4*)g, (AS3 i32x4*)l, 0, 0);
#else
    *(unsigned long long*)l       = *(const unsigned long long*)g;
    *(unsigned long long*)(l + 4) = *(const unsigned long long*)(g + 4);
#endif
}
// Threshold waits: leave up to N async copies (the next tile's group) in flight.
__device__ __forceinline__ void cp_wait0() {
#if HAVE_ASYNC_LDS
    asm volatile("s_wait_asynccnt 0x0" ::: "memory");
#endif
}
__device__ __forceinline__ void cp_wait2() {
#if HAVE_ASYNC_LDS
    asm volatile("s_wait_asynccnt 0x2" ::: "memory");
#endif
}
__device__ __forceinline__ void cp_wait4() {
#if HAVE_ASYNC_LDS
    asm volatile("s_wait_asynccnt 0x4" ::: "memory");
#endif
}

// ---------------------------------------------------------------------------
// Kernel 0: elementwise f32 -> bf16 (inputs and weights converted once).
// ---------------------------------------------------------------------------
__global__ __launch_bounds__(256)
void cvt_bf16_kernel(const float* __restrict__ src, unsigned short* __restrict__ dst, int n4)
{
    int i = blockIdx.x * blockDim.x + threadIdx.x;
    if (i < n4) {
        float4 f = ((const float4*)src)[i];
        ((unsigned long long*)dst)[i] = pack4_bf16(f);
    }
}

// ---------------------------------------------------------------------------
// Kernel 1: C = A(bf16)[8192x1024] @ W(bf16)[1024x1024] + bias.
// Block: 128x128 C tile, 256 thr (8 waves), wave = 32x64 (2x4 accums),
// K-step 32, double-buffered async LDS staging, peeled final tile.
// mode 0: bf16 out, head-split [B,H,S,HD]        (Q, V projections)
// mode 1: bf16 out, head-split+transposed [B,H,HD,S]  (K projection)
// mode 2: f32 out, [M,D] row-major               (output projection)
// ---------------------------------------------------------------------------
#define LA_STRIDE 40    // 128x32 bf16 A tile (80B rows: 16B-aligned, low conflict)
#define LB_STRIDE 136   // 32x128 bf16 B tile (272B rows)

__global__ __launch_bounds__(256)
void gemm_bf16_kernel(const unsigned short* __restrict__ A,
                      const unsigned short* __restrict__ W,
                      const float* __restrict__ bias,
                      void* __restrict__ outp, int mode)
{
    __shared__ unsigned short lA[2][128 * LA_STRIDE];
    __shared__ unsigned short lB[2][32 * LB_STRIDE];

    const int t    = threadIdx.x;
    const int wave = t >> 5, lane = t & 31, lr = lane & 15, half = lane >> 4;
    const int wm = wave & 3, wn = wave >> 2;            // 4x2 wave grid
    const int rowBase = blockIdx.y * 128;
    const int colBase = blockIdx.x * 128;

    // staging assignments: 512 16B chunks per tile, 2 per thread
    const int ar0 = (2 * t) >> 2,       ae0 = ((2 * t) & 3) * 8;
    const int ar1 = (2 * t + 1) >> 2,   ae1 = ((2 * t + 1) & 3) * 8;
    const int br0 = (2 * t) >> 4,       be0 = ((2 * t) & 15) * 8;
    const int br1 = (2 * t + 1) >> 4,   be1 = ((2 * t + 1) & 15) * 8;

    const unsigned short* gA0 = A + (long)(rowBase + ar0) * D_ + ae0;
    const unsigned short* gA1 = A + (long)(rowBase + ar1) * D_ + ae1;
    const unsigned short* gB0 = W + (long)br0 * D_ + colBase + be0;
    const unsigned short* gB1 = W + (long)br1 * D_ + colBase + be1;

    f32x8 acc[2][4] = {};

    auto stage = [&](int buf, int kt) {
        cp16_g2s(&lA[buf][ar0 * LA_STRIDE + ae0], gA0 + kt);
        cp16_g2s(&lA[buf][ar1 * LA_STRIDE + ae1], gA1 + kt);
        cp16_g2s(&lB[buf][br0 * LB_STRIDE + be0], gB0 + (long)kt * D_);
        cp16_g2s(&lB[buf][br1 * LB_STRIDE + be1], gB1 + (long)kt * D_);
    };
    auto compute = [&](int buf) {
        BFrag af[2], bf[4];
        for (int mt = 0; mt < 2; ++mt)
            for (int j = 0; j < 8; ++j)
                af[mt].u[j] = *(const unsigned int*)
                    &lA[buf][(wm * 32 + mt * 16 + lr) * LA_STRIDE + kpair(j, half)];
        for (int nt = 0; nt < 4; ++nt)
            for (int j = 0; j < 8; ++j)   // B 32x16: lane = N col, VGPR packs K pair
                bf[nt].u[j] = (unsigned int)lB[buf][kpair(j, half) * LB_STRIDE + wn * 64 + nt * 16 + lr]
                    | ((unsigned int)lB[buf][(kpair(j, half) + 1) * LB_STRIDE + wn * 64 + nt * 16 + lr] << 16);
        for (int mt = 0; mt < 2; ++mt)
            for (int nt = 0; nt < 4; ++nt)
                acc[mt][nt] = wmma_bf16(af[mt], bf[nt], acc[mt][nt]);
    };

    // pipeline: stage 0; steady-state loop is branch-free; final tile peeled
    stage(0, 0);
    int buf = 0;
    for (int kt = 0; kt + 32 < D_; kt += 32, buf ^= 1) {
        stage(buf ^ 1, kt + 32);
        cp_wait4();                      // tile kt complete; kt+32 in flight
        if (kt + 64 < D_) {              // prefetch two tiles ahead
            __builtin_prefetch(gA0 + kt + 64, 0, 3);
            __builtin_prefetch(gB0 + (long)(kt + 64) * D_, 0, 3);
        }
        __syncthreads();                 // tile kt visible to all waves
        compute(buf);
        __syncthreads();                 // all waves done reading buf
    }
    cp_wait0();
    __syncthreads();
    compute(buf);                        // final tile

    if (mode == 2) {
        float* o = (float*)outp;
        for (int mt = 0; mt < 2; ++mt)
            for (int nt = 0; nt < 4; ++nt)
                for (int r = 0; r < 8; ++r) {
                    int rg = rowBase + wm * 32 + mt * 16 + r + 8 * half;
                    int cg = colBase + wn * 64 + nt * 16 + lr;
                    o[(long)rg * D_ + cg] = acc[mt][nt][r] + bias[cg];
                }
    } else if (mode == 1) {
        // transposed head-split store: lane holds 8 consecutive s values
        unsigned short* o = (unsigned short*)outp;
        for (int mt = 0; mt < 2; ++mt)
            for (int nt = 0; nt < 4; ++nt) {
                int d  = colBase + wn * 64 + nt * 16 + lr;     // feature (HD dim)
                int sg = rowBase + wm * 32 + mt * 16 + 8 * half;
                int bb = sg >> 11, s = sg & (S_ - 1);
                int hh = d >> 6,   hd = d & (HD_ - 1);
                float bi = bias[d];
                f32x8 v = acc[mt][nt];
                unsigned long long lo =
                      (unsigned long long)pack2_bf16(v[0] + bi, v[1] + bi)
                    | ((unsigned long long)pack2_bf16(v[2] + bi, v[3] + bi) << 32);
                unsigned long long hi =
                      (unsigned long long)pack2_bf16(v[4] + bi, v[5] + bi)
                    | ((unsigned long long)pack2_bf16(v[6] + bi, v[7] + bi) << 32);
                unsigned short* dst = o + (((long)bb * H_ + hh) * HD_ + hd) * S_ + s;
                *(unsigned long long*)dst       = lo;
                *(unsigned long long*)(dst + 4) = hi;
            }
    } else {
        unsigned short* o = (unsigned short*)outp;
        for (int mt = 0; mt < 2; ++mt)
            for (int nt = 0; nt < 4; ++nt)
                for (int r = 0; r < 8; ++r) {
                    int rg = rowBase + wm * 32 + mt * 16 + r + 8 * half;
                    int cg = colBase + wn * 64 + nt * 16 + lr;
                    int bb = rg >> 11, s = rg & (S_ - 1);
                    int hh = cg >> 6,  hd = cg & (HD_ - 1);
                    o[(((long)bb * H_ + hh) * S_ + s) * HD_ + hd] =
                        cvt1_bf16(acc[mt][nt][r] + bias[cg]);
                }
    }
}

// ---------------------------------------------------------------------------
// Kernel 2: flash attention per (b,h).  Block = 128 q rows, 8 waves, each wave
// owns one 16-row tile, online softmax over keys in steps of 32, with a
// double-buffered async K/V tile pipeline (final tile peeled).
// Q: [B,H,S,HD]  K: [B,H,HD,S] (pre-transposed)  V: [B,H,S,HD]
// ---------------------------------------------------------------------------
#define KT_STRIDE 40   // KT[d][key] 64x32 bf16 (80B rows)
#define V_STRIDE  72   // V[key][d]  32x64 bf16 (144B rows)

__global__ __launch_bounds__(256)
void attn_kernel(const unsigned short* __restrict__ Q,
                 const unsigned short* __restrict__ Kt,
                 const unsigned short* __restrict__ Vm,
                 const float* __restrict__ mask,
                 unsigned short* __restrict__ ctx)
{
    __shared__ unsigned short lKT[2][64 * KT_STRIDE];
    __shared__ unsigned short lV[2][32 * V_STRIDE];
    __shared__ unsigned short lP[8][16 * 32];   // wave-private P tiles

    const int t    = threadIdx.x;
    const int wave = t >> 5, lane = t & 31, lr = lane & 15, half = lane >> 4;
    const int h = blockIdx.y, b = blockIdx.z;
    const long base = ((long)b * H_ + h) * S_ * HD_;   // same extent for both layouts
    const int q0 = blockIdx.x * 128 + wave * 16;

    // staging assignments (16B per thread per tile)
    const int kd  = t >> 2, kq = (t & 3) * 8;   // KT: d row, key offset
    const int vk  = t >> 3, vd = (t & 7) * 8;   // V: key row, d offset
    const unsigned short* gK = Kt + base + (long)kd * S_ + kq;
    const unsigned short* gV = Vm + base + (long)vk * HD_ + vd;

    // Q A-fragments (16 rows x 64 d = two 16x32 fragments), loaded once
    BFrag qf[2];
    for (int c = 0; c < 2; ++c)
        for (int j = 0; j < 8; ++j)
            qf[c].u[j] = *(const unsigned int*)
                (Q + base + (long)(q0 + lr) * HD_ + c * 32 + kpair(j, half));

    float mrow[8], lrow[8];
    f32x8 O[4] = {};
    for (int r = 0; r < 8; ++r) { mrow[r] = -1e30f; lrow[r] = 0.0f; }

    const float scale = 0.125f;        // 1/sqrt(64)
    const float NEGBIG = -1e9f;

    auto stage = [&](int buf, int kb0) {
        cp16_g2s(&lKT[buf][kd * KT_STRIDE + kq], gK + kb0);
        cp16_g2s(&lV[buf][vk * V_STRIDE + vd],   gV + (long)kb0 * HD_);
    };
    auto compute = [&](int buf, int kb0) {
        // scores: two 16x16 key tiles, each accumulated over 2 d-chunks
        f32x8 s0 = {}, s1 = {};
        BFrag kf;
        for (int c = 0; c < 2; ++c) {
            for (int j = 0; j < 8; ++j)
                kf.u[j] = (unsigned int)lKT[buf][(c * 32 + kpair(j, half)) * KT_STRIDE + lr]
                    | ((unsigned int)lKT[buf][(c * 32 + kpair(j, half) + 1) * KT_STRIDE + lr] << 16);
            s0 = wmma_bf16(qf[c], kf, s0);
        }
        for (int c = 0; c < 2; ++c) {
            for (int j = 0; j < 8; ++j)
                kf.u[j] = (unsigned int)lKT[buf][(c * 32 + kpair(j, half)) * KT_STRIDE + 16 + lr]
                    | ((unsigned int)lKT[buf][(c * 32 + kpair(j, half) + 1) * KT_STRIDE + 16 + lr] << 16);
            s1 = wmma_bf16(qf[c], kf, s1);
        }

        // scale + mask + online softmax (row stats per lane for its half)
        float corr[8];
        for (int r = 0; r < 8; ++r) {
            int q = q0 + r + 8 * half;
            const float* mp = mask + ((long)b * S_ + q) * S_ + kb0;
            float x0 = fmaf(s0[r], scale, mp[lr]      * NEGBIG);
            float x1 = fmaf(s1[r], scale, mp[16 + lr] * NEGBIG);
            float a = fmaxf(x0, x1);
            for (int off = 1; off < 16; off <<= 1)
                a = fmaxf(a, __shfl_xor(a, off, 32));   // stays within 16-lane half
            float mn = fmaxf(mrow[r], a);
            float cr = __expf(mrow[r] - mn);
            mrow[r] = mn;
            float p0 = __expf(x0 - mn), p1 = __expf(x1 - mn);
            s0[r] = p0; s1[r] = p1;
            float ls = p0 + p1;
            for (int off = 1; off < 16; off <<= 1)
                ls += __shfl_xor(ls, off, 32);
            lrow[r] = lrow[r] * cr + ls;
            corr[r] = cr;
        }
        for (int nt = 0; nt < 4; ++nt)
            for (int r = 0; r < 8; ++r)
                O[nt][r] *= corr[r];

        // P: C-layout f32 -> bf16 A-layout via wave-private LDS
        unsigned short* P = lP[wave];
        for (int r = 0; r < 8; ++r) {
            int row = r + 8 * half;
            P[row * 32 + lr]      = cvt1_bf16(s0[r]);
            P[row * 32 + 16 + lr] = cvt1_bf16(s1[r]);
        }
        asm volatile("s_wait_dscnt 0x0" ::: "memory");   // wave-local LDS RAW

        BFrag pf;
        for (int j = 0; j < 8; ++j)
            pf.u[j] = *(const unsigned int*)&P[lr * 32 + kpair(j, half)];

        for (int nt = 0; nt < 4; ++nt) {
            BFrag vf;
            for (int j = 0; j < 8; ++j)   // lane = d col, VGPR packs key pair
                vf.u[j] = (unsigned int)lV[buf][kpair(j, half) * V_STRIDE + nt * 16 + lr]
                    | ((unsigned int)lV[buf][(kpair(j, half) + 1) * V_STRIDE + nt * 16 + lr] << 16);
            O[nt] = wmma_bf16(pf, vf, O[nt]);
        }
    };

    // pipeline: stage 0; steady-state loop is branch-free; final tile peeled
    stage(0, 0);
    int buf = 0, kb0 = 0;
    for (; kb0 + 32 < S_; kb0 += 32, buf ^= 1) {
        stage(buf ^ 1, kb0 + 32);
        cp_wait2();                      // tile kb0 done; kb0+32 in flight
        if (kb0 + 64 < S_) {
            __builtin_prefetch(gK + kb0 + 64, 0, 3);
            __builtin_prefetch(gV + (long)(kb0 + 64) * HD_, 0, 3);
        }
        __syncthreads();
        compute(buf, kb0);
        __syncthreads();                 // all waves done reading buf
    }
    cp_wait0();
    __syncthreads();
    compute(buf, kb0);                   // final key tile

    // epilogue: normalize, bf16 store to context [B,S,D] (D index = h*64+d)
    for (int r = 0; r < 8; ++r) {
        float inv = 1.0f / lrow[r];
        int q = q0 + r + 8 * half;
        for (int nt = 0; nt < 4; ++nt) {
            int d = nt * 16 + lr;
            ctx[((long)b * S_ + q) * D_ + h * HD_ + d] = cvt1_bf16(O[nt][r] * inv);
        }
    }
}

// ---------------------------------------------------------------------------
extern "C" void kernel_launch(void* const* d_in, const int* in_sizes, int n_in,
                              void* d_out, int out_size, void* d_ws, size_t ws_size,
                              hipStream_t stream) {
    const float* query = (const float*)d_in[0];
    const float* key   = (const float*)d_in[1];
    const float* value = (const float*)d_in[2];
    const float* mask  = (const float*)d_in[3];
    const float* Wq = (const float*)d_in[4];
    const float* bq = (const float*)d_in[5];
    const float* Wk = (const float*)d_in[6];
    const float* bk = (const float*)d_in[7];
    const float* Wv = (const float*)d_in[8];
    const float* bv = (const float*)d_in[9];
    const float* Wo = (const float*)d_in[10];
    const float* bo = (const float*)d_in[11];

    const size_t elems = (size_t)M_ * D_;       // 8M
    const size_t welems = (size_t)D_ * D_;      // 1M
    unsigned short* p = (unsigned short*)d_ws;
    unsigned short* xq = p;  p += elems;        // bf16 copies of inputs
    unsigned short* xk = p;  p += elems;
    unsigned short* xv = p;  p += elems;
    unsigned short* wq = p;  p += welems;       // bf16 copies of weights
    unsigned short* wk = p;  p += welems;
    unsigned short* wv = p;  p += welems;
    unsigned short* wo = p;  p += welems;
    unsigned short* qb = p;  p += elems;        // projections
    unsigned short* kb = p;  p += elems;        // [B,H,HD,S] (transposed)
    unsigned short* vb = p;  p += elems;
    unsigned short* cb = p;  p += elems;        // context [B,S,D]

    const int n4i = (int)(elems / 4), n4w = (int)(welems / 4);
    cvt_bf16_kernel<<<(n4i + 255) / 256, 256, 0, stream>>>(query, xq, n4i);
    cvt_bf16_kernel<<<(n4i + 255) / 256, 256, 0, stream>>>(key,   xk, n4i);
    cvt_bf16_kernel<<<(n4i + 255) / 256, 256, 0, stream>>>(value, xv, n4i);
    cvt_bf16_kernel<<<(n4w + 255) / 256, 256, 0, stream>>>(Wq, wq, n4w);
    cvt_bf16_kernel<<<(n4w + 255) / 256, 256, 0, stream>>>(Wk, wk, n4w);
    cvt_bf16_kernel<<<(n4w + 255) / 256, 256, 0, stream>>>(Wv, wv, n4w);
    cvt_bf16_kernel<<<(n4w + 255) / 256, 256, 0, stream>>>(Wo, wo, n4w);

    dim3 gGrid(D_ / 128, M_ / 128);   // (8, 64)
    gemm_bf16_kernel<<<gGrid, 256, 0, stream>>>(xq, wq, bq, qb, 0);
    gemm_bf16_kernel<<<gGrid, 256, 0, stream>>>(xk, wk, bk, kb, 1);
    gemm_bf16_kernel<<<gGrid, 256, 0, stream>>>(xv, wv, bv, vb, 0);

    attn_kernel<<<dim3(S_ / 128, H_, B_), 256, 0, stream>>>(qb, kb, vb, mask, cb);

    gemm_bf16_kernel<<<gGrid, 256, 0, stream>>>(cb, wo, bo, d_out, 2);
}
